// SwinTransformerBlock_41205916238072
// MI455X (gfx1250) — matrix operand order repaired and checked by reading, not compiled, not run on hardware
//
#include <hip/hip_runtime.h>

typedef __attribute__((ext_vector_type(8)))  __bf16 v8bf;
typedef __attribute__((ext_vector_type(16))) __bf16 v16bf;
typedef __attribute__((ext_vector_type(8)))  float  v8f;

#define WMMA_BF16(a, b, c) \
  __builtin_amdgcn_wmma_f32_16x16x32_bf16(false, (a), false, (b), (short)0, (c), false, false)

// ---------------------------------------------------------------------------
// Fragment loaders (wave32; A/B use this file's contiguous k order, same for both)
// A 16x32 bf16: lane<16 holds row M=lane, K=0..15 contiguous;
//               lane>=16 row M=lane-16, K=16..31 contiguous.
// B 32x16 bf16: lane<16 holds col N=lane, K=0..15 contiguous;
//               lane>=16 col N=lane-16, K=16..31 contiguous.
// C/D 16x16 f32: lane holds col N=lane&15, VGPR r = row M = r + 8*(lane>=16).
// ---------------------------------------------------------------------------
__device__ __forceinline__ v16bf frag_a(const __bf16* rowptr, int half) {
  const __bf16* q = rowptr + half * 16;
  v8bf c0 = *(const v8bf*)(q);
  v8bf c1 = *(const v8bf*)(q + 8);
  v16bf r;
#pragma unroll
  for (int i = 0; i < 8; ++i) { r[i] = c0[i]; r[i + 8] = c1[i]; }
  return r;
}
__device__ __forceinline__ v16bf frag_b(const __bf16* colptr, int half) {
  return *(const v16bf*)(colptr + half * 16);
}
// B-frag from LDS with 16B-granular loads (col stride padded, 16B aligned)
__device__ __forceinline__ v16bf frag_b_lds(const __bf16* colptr, int half) {
  const __bf16* p = colptr + half * 16;
  v8bf b0 = *(const v8bf*)(p);
  v8bf b1 = *(const v8bf*)(p + 8);
  v16bf r;
#pragma unroll
  for (int i = 0; i < 8; ++i) { r[i] = b0[i]; r[i + 8] = b1[i]; }
  return r;
}

// ---------------------------------------------------------------------------
// Constants
// ---------------------------------------------------------------------------
#define BB 8
#define HH 96
#define WW_ 288
#define CC 256
#define NHEAD 8
#define HD 32
#define NTOK 27
#define NWIN_ALL 8192         // B * 1024 windows
#define MTOT 221184           // NWIN_ALL * 27 == B*H*W
#define SCALE 0.17677669529663687f
#define BSTRIDE 40            // LDS col stride in bf16 elems (80B, bank-conflict-free)

// ---------------------------------------------------------------------------
// float -> bf16 weight conversion
// ---------------------------------------------------------------------------
__global__ void cvt_bf16(const float* __restrict__ s, __bf16* __restrict__ d, int n) {
  int i = blockIdx.x * 256 + threadIdx.x;
  if (i < n) d[i] = (__bf16)s[i];
}

// ---------------------------------------------------------------------------
// LN1 + cyclic shift(-1,-1) + window partition -> bf16 windowed tokens
// ---------------------------------------------------------------------------
__global__ __launch_bounds__(256) void ln_win_kernel(
    const float* __restrict__ x, const float* __restrict__ g,
    const float* __restrict__ b, __bf16* __restrict__ out) {
  const int lane = threadIdx.x & 31, wid = threadIdx.x >> 5;
  const int wt = blockIdx.x * 8 + wid;            // windowed token index
  const int win = wt / NTOK, n = wt % NTOK;
  const int b_ = win >> 10, wi = win & 1023;
  const int hsh = (wi >> 5) * 3 + n / 9;
  const int wsh = (wi & 31) * 9 + n % 9;
  int h = hsh + 1; if (h == HH) h = 0;            // roll(-1): src = (pos+1)%H
  int w = wsh + 1; if (w == WW_) w = 0;
  const float* src = x + ((size_t)b_ * (HH * WW_) + (size_t)h * WW_ + w) * CC;
  float v[8], s = 0.f;
#pragma unroll
  for (int i = 0; i < 8; ++i) { v[i] = src[i * 32 + lane]; s += v[i]; }
#pragma unroll
  for (int d = 1; d < 32; d <<= 1) s += __shfl_xor(s, d);
  const float mean = s * (1.f / 256.f);
  float var = 0.f;
#pragma unroll
  for (int i = 0; i < 8; ++i) { float d0 = v[i] - mean; var += d0 * d0; }
#pragma unroll
  for (int d = 1; d < 32; d <<= 1) var += __shfl_xor(var, d);
  const float rs = rsqrtf(var * (1.f / 256.f) + 1e-5f);
  __bf16* dst = out + (size_t)wt * CC;
#pragma unroll
  for (int i = 0; i < 8; ++i) {
    int c = i * 32 + lane;
    dst[c] = (__bf16)((v[i] - mean) * rs * g[c] + b[c]);
  }
}

// LN2 over natural tokens (fp32 in -> bf16 out)
__global__ __launch_bounds__(256) void ln_plain_kernel(
    const float* __restrict__ x, const float* __restrict__ g,
    const float* __restrict__ b, __bf16* __restrict__ out) {
  const int lane = threadIdx.x & 31, wid = threadIdx.x >> 5;
  const int tok = blockIdx.x * 8 + wid;
  const float* src = x + (size_t)tok * CC;
  float v[8], s = 0.f;
#pragma unroll
  for (int i = 0; i < 8; ++i) { v[i] = src[i * 32 + lane]; s += v[i]; }
#pragma unroll
  for (int d = 1; d < 32; d <<= 1) s += __shfl_xor(s, d);
  const float mean = s * (1.f / 256.f);
  float var = 0.f;
#pragma unroll
  for (int i = 0; i < 8; ++i) { float d0 = v[i] - mean; var += d0 * d0; }
#pragma unroll
  for (int d = 1; d < 32; d <<= 1) var += __shfl_xor(var, d);
  const float rs = rsqrtf(var * (1.f / 256.f) + 1e-5f);
  __bf16* dst = out + (size_t)tok * CC;
#pragma unroll
  for (int i = 0; i < 8; ++i) {
    int c = i * 32 + lane;
    dst[c] = (__bf16)((v[i] - mean) * rs * g[c] + b[c]);
  }
}

// ---------------------------------------------------------------------------
// WMMA GEMM: out[M,N] = A[M,K] * Wt[N,K]^T + bias, fused epilogues.
// Block 256 thr = 8 waves (2x4), block tile 64x256, wave tile 32x64.
// B tiles (256 cols x 32 k, 16KB) staged via double-buffered
// global_load_async_to_lds_b128 (ASYNCcnt pipeline); A direct from global.
// EPI: 0 = store bf16 | 1 = GELU(exact)+bf16 | 2 = proj: window-reverse +
//      roll(+1,+1) scatter + shortcut add -> f32 | 3 = residual add -> f32
// ---------------------------------------------------------------------------
template <int K, int EPI>
__global__ __launch_bounds__(256) void gemm_wmma(
    const __bf16* __restrict__ A, const __bf16* __restrict__ Wt,
    const float* __restrict__ bias, void* __restrict__ outp,
    const float* __restrict__ resid, int Nout) {
  constexpr int NK = K / 32;
  __shared__ __bf16 btile[2][256 * BSTRIDE];   // 2 x 20KB double buffer

  const int tid  = threadIdx.x;
  const int lane = tid & 31;
  const int wid  = tid >> 5;
  const int half = lane >> 4;
  const int l16  = lane & 15;
  const int mWave = blockIdx.x * 64 + (wid >> 2) * 32;
  const int nWave = blockIdx.y * 256 + (wid & 3) * 64;

  v8f acc[2][4];
#pragma unroll
  for (int mi = 0; mi < 2; ++mi)
#pragma unroll
    for (int ni = 0; ni < 4; ++ni) acc[mi][ni] = (v8f)(0.f);

  const __bf16* arow0 = A + (size_t)(mWave + l16) * K;
  const __bf16* arow1 = A + (size_t)(mWave + 16 + l16) * K;

  // this thread stages one weight column (64B per k-step = 4x b128)
  const __bf16* wcol = Wt + (size_t)(blockIdx.y * 256 + tid) * K;
  const uint32_t lds0 = (uint32_t)(uintptr_t)(&btile[0][0]) + tid * (BSTRIDE * 2);
  const uint32_t lds1 = (uint32_t)(uintptr_t)(&btile[1][0]) + tid * (BSTRIDE * 2);

  auto issue_tile = [&](uint32_t ldsbase, int k) {
    const __bf16* g = wcol + k;
#pragma unroll
    for (int i = 0; i < 4; ++i) {
      asm volatile("global_load_async_to_lds_b128 %0, %1, off"
                   :: "v"(ldsbase + i * 16),
                      "v"((unsigned long long)(uintptr_t)(g + i * 8))
                   : "memory");
    }
  };

  issue_tile(lds0, 0);
  for (int ks = 0; ks < NK; ++ks) {
    if (ks + 1 < NK) {
      issue_tile((ks & 1) ? lds0 : lds1, (ks + 1) * 32);
      asm volatile("s_wait_asynccnt 0x4" ::: "memory");  // current tile landed
    } else {
      asm volatile("s_wait_asynccnt 0x0" ::: "memory");
    }
    __syncthreads();

    const __bf16* bbuf = (ks & 1) ? &btile[1][0] : &btile[0][0];
    v16bf a0 = frag_a(arow0 + ks * 32, half);
    v16bf a1 = frag_a(arow1 + ks * 32, half);
#pragma unroll
    for (int ni = 0; ni < 4; ++ni) {
      const __bf16* bp = bbuf + ((wid & 3) * 64 + ni * 16 + l16) * BSTRIDE;
      v16bf bf = frag_b_lds(bp, half);
      acc[0][ni] = WMMA_BF16(a0, bf, acc[0][ni]);
      acc[1][ni] = WMMA_BF16(a1, bf, acc[1][ni]);
    }
    __syncthreads();  // all waves done with this buffer before it refills
  }

#pragma unroll
  for (int ni = 0; ni < 4; ++ni) {
    const int col = nWave + ni * 16 + l16;
    const float bv = bias[col];
#pragma unroll
    for (int mi = 0; mi < 2; ++mi) {
#pragma unroll
      for (int r = 0; r < 8; ++r) {
        const int m = mWave + mi * 16 + r + half * 8;
        float v = acc[mi][ni][r] + bv;
        if constexpr (EPI == 0) {
          ((__bf16*)outp)[(size_t)m * Nout + col] = (__bf16)v;
        } else if constexpr (EPI == 1) {
          v = 0.5f * v * (1.f + erff(v * 0.70710678118654752f));
          ((__bf16*)outp)[(size_t)m * Nout + col] = (__bf16)v;
        } else if constexpr (EPI == 2) {
          // m is windowed row: win*27+n -> natural token (reverse + roll +1)
          const int win = m / NTOK, n = m % NTOK;
          const int b_ = win >> 10, wi = win & 1023;
          int h = (wi >> 5) * 3 + n / 9 + 1; if (h == HH) h = 0;
          int w = (wi & 31) * 9 + n % 9 + 1; if (w == WW_) w = 0;
          const size_t tok = (size_t)b_ * (HH * WW_) + (size_t)h * WW_ + w;
          ((float*)outp)[tok * CC + col] = resid[tok * CC + col] + v;
        } else {
          ((float*)outp)[(size_t)m * CC + col] = resid[(size_t)m * CC + col] + v;
        }
      }
    }
  }
}

// ---------------------------------------------------------------------------
// Windowed attention: one wave per (window, head).
// S = Q K^T (4 wmma) -> scale + rel-pos bias + shift mask -> softmax (shfl)
// -> P staged bf16 in LDS, V staged transposed in LDS -> O = P V (4 wmma)
// ---------------------------------------------------------------------------
__global__ __launch_bounds__(256) void attn_kernel(
    const __bf16* __restrict__ qkv, const float* __restrict__ rpb,
    __bf16* __restrict__ out) {
  __shared__ __bf16 vt_lds[8][1024];   // V^T per wave: [hd(32)][tok(32)]
  __shared__ __bf16 p_lds[8][1024];    // P per wave:   [i(32)][j(32)]
  const int lane = threadIdx.x & 31;
  const int wid  = threadIdx.x >> 5;
  const int half = lane >> 4;
  const int l16  = lane & 15;
  const int task = blockIdx.x * 8 + wid;
  const int win  = task >> 3;
  const int head = task & 7;
  const size_t rowBase = (size_t)win * NTOK;

  __bf16* vt = vt_lds[wid];
  __bf16* pb = p_lds[wid];

  // stage V transposed (pad tokens 27..31 with zeros)
  {
    const int t = lane;
    if (t < NTOK) {
      const __bf16* src = qkv + (rowBase + t) * 768 + 512 + head * HD;
      v16bf x0 = *(const v16bf*)(src);
      v16bf x1 = *(const v16bf*)(src + 16);
#pragma unroll
      for (int hd = 0; hd < 16; ++hd) vt[hd * 32 + t] = x0[hd];
#pragma unroll
      for (int hd = 0; hd < 16; ++hd) vt[(hd + 16) * 32 + t] = x1[hd];
    } else {
#pragma unroll
      for (int hd = 0; hd < 32; ++hd) vt[hd * 32 + t] = (__bf16)0.f;
    }
  }

  // S = Q K^T  (K-dim = head_dim = 32 -> one wmma per 16x16 tile)
  v8f s[2][2];
#pragma unroll
  for (int mi = 0; mi < 2; ++mi)
#pragma unroll
    for (int ni = 0; ni < 2; ++ni) s[mi][ni] = (v8f)(0.f);

  const __bf16* qbase = qkv + head * HD;        // q cols [0,256)
  const __bf16* kbase = qkv + 256 + head * HD;  // k cols [256,512)
#pragma unroll
  for (int mi = 0; mi < 2; ++mi) {
    int i = mi * 16 + l16; if (i > 26) i = 26;  // clamp padded rows
    v16bf a = frag_a(qbase + (rowBase + i) * 768, half);
#pragma unroll
    for (int ni = 0; ni < 2; ++ni) {
      int j = ni * 16 + l16; if (j > 26) j = 26; // clamp padded cols
      v16bf bf = frag_b(kbase + (rowBase + j) * 768, half);
      s[mi][ni] = WMMA_BF16(a, bf, s[mi][ni]);
    }
  }

  // bias + shift mask + softmax; write normalized P (bf16) into LDS
  const int wi = win & 1023;
  const int hb = (wi >> 5) * 3, wb = (wi & 31) * 9;
#pragma unroll
  for (int mi = 0; mi < 2; ++mi) {
#pragma unroll
    for (int r = 0; r < 8; ++r) {
      const int i = mi * 16 + r + half * 8;
      const int ai = i / 9, bi = i % 9;
      const int hi = hb + ai, wic = wb + bi;
      const int li = (hi < 93 ? 0 : (hi < 95 ? 1 : 2)) * 3 +
                     (wic < 279 ? 0 : (wic < 284 ? 1 : 2));
      float vv[2];
#pragma unroll
      for (int ni = 0; ni < 2; ++ni) {
        const int j = ni * 16 + l16;
        float v = s[mi][ni][r] * SCALE;
        if (i < NTOK && j < NTOK) {
          const int aj = j / 9, bj = j % 9;
          const int idx = (ai - aj + 2) * 11 + (bi - bj + 8);
          v += rpb[idx * NHEAD + head];
          const int hj = hb + aj, wjc = wb + bj;
          const int lj = (hj < 93 ? 0 : (hj < 95 ? 1 : 2)) * 3 +
                         (wjc < 279 ? 0 : (wjc < 284 ? 1 : 2));
          if (li != lj) v -= 100.f;
        } else {
          v = -1e30f;
        }
        vv[ni] = v;
      }
      float mx = fmaxf(vv[0], vv[1]);
#pragma unroll
      for (int d = 1; d <= 8; d <<= 1) mx = fmaxf(mx, __shfl_xor(mx, d));
      const float e0 = __expf(vv[0] - mx), e1 = __expf(vv[1] - mx);
      float ssum = e0 + e1;
#pragma unroll
      for (int d = 1; d <= 8; d <<= 1) ssum += __shfl_xor(ssum, d);
      const float inv = 1.f / ssum;
      pb[i * 32 + l16]      = (__bf16)(e0 * inv);
      pb[i * 32 + 16 + l16] = (__bf16)(e1 * inv);
    }
  }
  __syncthreads();  // make cross-lane LDS writes (P, V^T) visible

  // O = P V : A-frags from p_lds, B-frags from vt_lds (both ld = 32)
  v8f o[2][2];
#pragma unroll
  for (int mi = 0; mi < 2; ++mi)
#pragma unroll
    for (int ni = 0; ni < 2; ++ni) o[mi][ni] = (v8f)(0.f);
#pragma unroll
  for (int mi = 0; mi < 2; ++mi) {
    v16bf a = frag_a(pb + (mi * 16 + l16) * 32, half);
#pragma unroll
    for (int ni = 0; ni < 2; ++ni) {
      v16bf bf = frag_b(vt + (ni * 16 + l16) * 32, half);
      o[mi][ni] = WMMA_BF16(a, bf, o[mi][ni]);
    }
  }

  // store: out[win*27+i][head*32 + hd] bf16
#pragma unroll
  for (int mi = 0; mi < 2; ++mi) {
#pragma unroll
    for (int ni = 0; ni < 2; ++ni) {
#pragma unroll
      for (int r = 0; r < 8; ++r) {
        const int i = mi * 16 + r + half * 8;
        if (i < NTOK)
          out[(rowBase + i) * CC + head * HD + ni * 16 + l16] =
              (__bf16)o[mi][ni][r];
      }
    }
  }
}

// ---------------------------------------------------------------------------
// Launcher
// ---------------------------------------------------------------------------
extern "C" void kernel_launch(void* const* d_in, const int* in_sizes, int n_in,
                              void* d_out, int out_size, void* d_ws, size_t ws_size,
                              hipStream_t stream) {
  const float* x      = (const float*)d_in[0];
  const float* n1g    = (const float*)d_in[1];
  const float* n1b    = (const float*)d_in[2];
  const float* qkv_w  = (const float*)d_in[3];
  const float* qkv_b  = (const float*)d_in[4];
  const float* rpb    = (const float*)d_in[5];
  const float* proj_w = (const float*)d_in[6];
  const float* proj_b = (const float*)d_in[7];
  const float* n2g    = (const float*)d_in[8];
  const float* n2b    = (const float*)d_in[9];
  const float* fc1_w  = (const float*)d_in[10];
  const float* fc1_b  = (const float*)d_in[11];
  const float* fc2_w  = (const float*)d_in[12];
  const float* fc2_b  = (const float*)d_in[13];

  char* ws = (char*)d_ws;
  __bf16* bufA = (__bf16*)(ws);                        // 113,246,208 B (xw / attn_out / xn2)
  __bf16* bufB = (__bf16*)(ws + 113246208ull);         // 452,984,832 B (qkv / mlp hidden)
  float*  xres = (float*) (ws + 566231040ull);         // 226,492,416 B (x + attn residual)
  __bf16* wq   = (__bf16*)(ws + 792723456ull);         // bf16 weights
  __bf16* wp   = wq  + 768 * 256;
  __bf16* wf1  = wp  + 256 * 256;
  __bf16* wf2  = wf1 + 1024 * 256;

  cvt_bf16<<<(768 * 256 + 255) / 256, 256, 0, stream>>>(qkv_w, wq, 768 * 256);
  cvt_bf16<<<(256 * 256 + 255) / 256, 256, 0, stream>>>(proj_w, wp, 256 * 256);
  cvt_bf16<<<(1024 * 256 + 255) / 256, 256, 0, stream>>>(fc1_w, wf1, 1024 * 256);
  cvt_bf16<<<(256 * 1024 + 255) / 256, 256, 0, stream>>>(fc2_w, wf2, 256 * 1024);

  // 1) LN1 + shift + window partition
  ln_win_kernel<<<MTOT / 8, 256, 0, stream>>>(x, n1g, n1b, bufA);
  // 2) QKV GEMM -> bf16 [M][768]
  gemm_wmma<256, 0><<<dim3(MTOT / 64, 3), 256, 0, stream>>>(bufA, wq, qkv_b, bufB, nullptr, 768);
  // 3) windowed attention -> bf16 [M][256] (overwrites bufA)
  attn_kernel<<<NWIN_ALL * NHEAD / 8, 256, 0, stream>>>(bufB, rpb, bufA);
  // 4) proj GEMM + window-reverse + roll + shortcut -> f32 xres
  gemm_wmma<256, 2><<<dim3(MTOT / 64, 1), 256, 0, stream>>>(bufA, wp, proj_b, xres, x, 256);
  // 5) LN2 -> bf16 (bufA)
  ln_plain_kernel<<<MTOT / 8, 256, 0, stream>>>(xres, n2g, n2b, bufA);
  // 6) FC1 + GELU -> bf16 [M][1024] (bufB)
  gemm_wmma<256, 1><<<dim3(MTOT / 64, 4), 256, 0, stream>>>(bufA, wf1, fc1_b, bufB, nullptr, 1024);
  // 7) FC2 + residual -> f32 d_out
  gemm_wmma<1024, 3><<<dim3(MTOT / 64, 1), 256, 0, stream>>>(bufB, wf2, fc2_b, d_out, xres, 256);
}
